// NaiveDynMessage_58248346468617
// MI455X (gfx1250) — compile-verified, matrix operand order checked by reading, not run on hardware
//
#include <hip/hip_runtime.h>
#include <math.h>

typedef __attribute__((ext_vector_type(16))) _Float16 v16h;
typedef __attribute__((ext_vector_type(8)))  _Float16 v8h;
typedef __attribute__((ext_vector_type(8)))  float    v8f;
typedef __attribute__((ext_vector_type(4)))  float    v4f;

#define N_V   2048
#define N_E   8192
#define TWO_E 16384
#define HV    256
#define HE    128
#define MV    256
#define ME    256
#define KLINK 544   // 518 padded to 17*32

__device__ __forceinline__ void atomicMaxF(float* addr, float val) {
    // monotone int trick: works for mixed-sign floats with -inf init
    if (val >= 0.f) atomicMax((int*)addr, __float_as_int(val));
    else            atomicMin((unsigned int*)addr, (unsigned int)__float_as_int(val));
}

// ---------------- init scratch ----------------
__global__ void k_init(float* mvAcc, float* nodeSum, int* nodeDeg,
                       float* nodeMax, float* colSum) {
    const int T0 = N_V * MV;
    const int TOT = T0 + 3 * N_V + MV;
    int stride = gridDim.x * blockDim.x;
    for (int i = blockIdx.x * blockDim.x + threadIdx.x; i < TOT; i += stride) {
        if (i < T0)                 mvAcc[i] = 0.f;
        else if (i < T0 + N_V)      nodeSum[i - T0] = 0.f;
        else if (i < T0 + 2 * N_V)  nodeDeg[i - T0 - N_V] = 0;
        else if (i < T0 + 3 * N_V)  nodeMax[i - T0 - 2 * N_V] = -INFINITY;
        else                        colSum[i - T0 - 3 * N_V] = 0.f;
    }
}

// ---------------- f16 weight packing (B stored [N][K], link rows permuted) --
__global__ void k_weights(const float* __restrict__ attend_w,
                          const float* __restrict__ link_w,
                          _Float16* __restrict__ awT, _Float16* __restrict__ lwT) {
    const int A = HV * MV;          // 65536
    const int L = MV * KLINK;       // 139264
    int stride = gridDim.x * blockDim.x;
    for (int i = blockIdx.x * blockDim.x + threadIdx.x; i < A + L; i += stride) {
        if (i < A) {
            int col = i >> 8, k = i & 255;
            awT[col * HV + k] = (_Float16)attend_w[k * MV + col];
        } else {
            int j = i - A;
            int col = j / KLINK, k = j - col * KLINK;
            float v = 0.f;
            if (k < 256)      v = link_w[(262 + k) * ME + col];        // hv_v
            else if (k < 512) v = link_w[(k - 256) * ME + col];        // hv_u
            else if (k < 515) v = link_w[(256 + k - 512) * ME + col];  // p
            else if (k < 518) v = link_w[(259 + k - 515) * ME + col];  // q
            lwT[col * KLINK + k] = (_Float16)v;
        }
    }
}

// -------- recover src/dst from one-hot vew: NT b128 streaming scan --------
__global__ void k_extract(const v4f* __restrict__ vew1,
                          const v4f* __restrict__ vew2,
                          int* __restrict__ uIdx, int* __restrict__ vIdx) {
    const long T4 = (long)N_V * N_E / 4;    // float4 elements per matrix
    long stride = (long)gridDim.x * blockDim.x;
    for (long i = blockIdx.x * (long)blockDim.x + threadIdx.x; i < 2 * T4; i += stride) {
        bool first = i < T4;
        long j = first ? i : i - T4;
        v4f x = __builtin_nontemporal_load(first ? vew1 + j : vew2 + j);
        long b = j * 4;                      // flat element index
        int n = (int)(b >> 13);              // row (E = 2^13)
        int e = (int)(b & (N_E - 1));        // col; e..e+3 stay in the same row
        #pragma unroll
        for (int c = 0; c < 4; c++) {
            if (x[c] > 0.5f) {
                int ec = e + c;
                if (first) { uIdx[ec] = n; vIdx[ec + N_E] = n; }   // src
                else       { vIdx[ec] = n; uIdx[ec + N_E] = n; }   // dst
            }
        }
    }
}

// ---------------- per-edge gather + align (wave per edge) ----------------
__global__ __launch_bounds__(256)
void k_edge(const float* __restrict__ hv_ftr, const float* __restrict__ he_ftr,
            const float* __restrict__ p_ftr,  const float* __restrict__ q_ftr,
            const float* __restrict__ align_w, const float* __restrict__ align_b,
            const int* __restrict__ uIdx, const int* __restrict__ vIdx,
            _Float16* __restrict__ mein, float* __restrict__ alignBuf,
            float* __restrict__ nodeMax, int* __restrict__ nodeDeg) {
    int tid = threadIdx.x, lane = tid & 31;
    int e = blockIdx.x * 8 + (tid >> 5);
    if (e >= TWO_E) return;
    int eu = uIdx[e], ev = vIdx[e];
    _Float16* row = mein + (size_t)e * KLINK;
    const float* hvv = hv_ftr + (size_t)ev * HV;
    const float* hvu = hv_ftr + (size_t)eu * HV;
    // each lane converts 8 contiguous f32 -> one b128 f16 store, twice
    int k0 = lane * 8;
    {
        v4f x0 = *(const v4f*)(hvv + k0);
        v4f x1 = *(const v4f*)(hvv + k0 + 4);
        v8h h;
        #pragma unroll
        for (int i = 0; i < 4; i++) { h[i] = (_Float16)x0[i]; h[4 + i] = (_Float16)x1[i]; }
        *(v8h*)(row + k0) = h;
        v4f y0 = *(const v4f*)(hvu + k0);
        v4f y1 = *(const v4f*)(hvu + k0 + 4);
        v8h g;
        #pragma unroll
        for (int i = 0; i < 4; i++) { g[i] = (_Float16)y0[i]; g[4 + i] = (_Float16)y1[i]; }
        *(v8h*)(row + HV + k0) = g;
    }
    float s = 0.f;
    if (lane < 3) {
        float d = p_ftr[ev * 3 + lane] - p_ftr[eu * 3 + lane];
        row[512 + lane] = (_Float16)d;
        s = d * align_w[lane];
    } else if (lane < 6) {
        int j = lane - 3;
        float d = q_ftr[ev * 3 + j] - q_ftr[eu * 3 + j];
        row[512 + lane] = (_Float16)d;
        s = d * align_w[lane];
    } else {
        row[512 + lane] = (_Float16)0.f;    // pad 518..543
    }
    const float* he = he_ftr + (size_t)(e < N_E ? e : e - N_E) * HE;
    for (int k = lane; k < HE; k += 32) s += he[k] * align_w[6 + k];
    for (int off = 16; off > 0; off >>= 1) s += __shfl_down(s, off, 32);
    if (lane == 0) {
        float a = s + align_b[0];
        alignBuf[e] = a;
        atomicMaxF(&nodeMax[ev], a);
        atomicAdd(&nodeDeg[ev], 1);
    }
}

// ------ WMMA GEMM, 2x2 register-blocked: 32x32 output per wave ------------
// out = leaky(A[16384,K] * B^T + bias), B stored [N][K]
__global__ __launch_bounds__(256)
void k_gemm(const _Float16* __restrict__ A, int lda,
            const _Float16* __restrict__ B, int K,
            const float* __restrict__ bias, float* __restrict__ out, int ntStore) {
    int tid = threadIdx.x;
    int wid = blockIdx.x * 8 + (tid >> 5);
    int tm = wid & 511;        // 512 M super-tiles of 32 rows
    int tn = wid >> 9;         // 8 N super-tiles of 32 cols
    if (tn >= MV / 32) return;
    int lane = tid & 31;
    int r  = lane & 15;
    int kh = (lane >> 4) << 3;
    const _Float16* A0 = A + (size_t)(tm * 32 + r) * lda;
    const _Float16* A1 = A0 + (size_t)16 * lda;
    const _Float16* B0 = B + (size_t)(tn * 32 + r) * K;
    const _Float16* B1 = B0 + (size_t)16 * K;
    v8f c00 = {}, c01 = {}, c10 = {}, c11 = {};
    auto frag = [](const _Float16* p) -> v16h {
        v8h x0 = *(const v8h*)(p);
        v8h x1 = *(const v8h*)(p + 16);
        return __builtin_shufflevector(x0, x1, 0,1,2,3,4,5,6,7,8,9,10,11,12,13,14,15);
    };
    for (int kb = 0; kb < K; kb += 32) {
        v16h a0 = frag(A0 + kb + kh);
        v16h a1 = frag(A1 + kb + kh);
        v16h b0 = frag(B0 + kb + kh);
        v16h b1 = frag(B1 + kb + kh);
        c00 = __builtin_amdgcn_wmma_f32_16x16x32_f16(false, a0, false, b0, (short)0, c00, false, false);
        c01 = __builtin_amdgcn_wmma_f32_16x16x32_f16(false, a0, false, b1, (short)0, c01, false, false);
        c10 = __builtin_amdgcn_wmma_f32_16x16x32_f16(false, a1, false, b0, (short)0, c10, false, false);
        c11 = __builtin_amdgcn_wmma_f32_16x16x32_f16(false, a1, false, b1, (short)0, c11, false, false);
    }
    int colBase = tn * 32 + r;
    int rowBase = tm * 32 + ((lane >> 4) << 3);
    auto store = [&](v8f c, int row0, int col) {
        float bv = bias[col];
        #pragma unroll
        for (int j = 0; j < 8; j++) {
            float v = c[j] + bv;
            v = v >= 0.f ? v : 0.01f * v;
            float* p = out + (size_t)(row0 + j) * MV + col;
            if (ntStore) __builtin_nontemporal_store(v, p); else *p = v;
        }
    };
    store(c00, rowBase,      colBase);
    store(c01, rowBase,      colBase + 16);
    store(c10, rowBase + 16, colBase);
    store(c11, rowBase + 16, colBase + 16);
}

// ---------------- segment softmax pieces ----------------
__global__ void k_expsum(const float* __restrict__ alignBuf, const int* __restrict__ vIdx,
                         const float* __restrict__ nodeMax,
                         float* __restrict__ expAlign, float* __restrict__ nodeSum) {
    int e = blockIdx.x * blockDim.x + threadIdx.x;
    if (e >= TWO_E) return;
    int v = vIdx[e];
    float w = expf(alignBuf[e] - nodeMax[v]);
    expAlign[e] = w;
    atomicAdd(&nodeSum[v], w);
}

__global__ void k_colsum(const float* __restrict__ attend, float* __restrict__ colSum) {
    int c = threadIdx.x;                        // 256 cols
    float s = 0.f;
    int e0 = blockIdx.x * (TWO_E / 64);
    for (int e = e0; e < e0 + TWO_E / 64; e++) s += attend[(size_t)e * MV + c];
    atomicAdd(&colSum[c], s);
}

__global__ __launch_bounds__(256)
void k_mvacc(const float* __restrict__ attend, const float* __restrict__ expAlign,
             const float* __restrict__ nodeSum, const int* __restrict__ vIdx,
             float* __restrict__ mvAcc) {
    int tid = threadIdx.x, lane = tid & 31;
    int e = blockIdx.x * 8 + (tid >> 5);
    if (e >= TWO_E) return;
    int v = vIdx[e];
    float w = expAlign[e] / nodeSum[v];
    const float* arow = attend + (size_t)e * MV;
    float* dst = mvAcc + (size_t)v * MV;
    for (int c = lane; c < MV; c += 32) atomicAdd(&dst[c], w * arow[c]);
}

__global__ void k_mvfin(const float* __restrict__ mvAcc, const int* __restrict__ nodeDeg,
                        const float* __restrict__ colSum, float* __restrict__ out) {
    int i = blockIdx.x * blockDim.x + threadIdx.x;
    if (i >= N_V * MV) return;
    int n = i >> 8, c = i & 255;
    float v = nodeDeg[n] ? mvAcc[i] : colSum[c] * (1.0f / TWO_E); // empty row -> uniform softmax
    out[i] = v > 0.f ? v : expm1f(v);                             // elu
}

extern "C" void kernel_launch(void* const* d_in, const int* in_sizes, int n_in,
                              void* d_out, int out_size, void* d_ws, size_t ws_size,
                              hipStream_t stream) {
    const float* hv_ftr   = (const float*)d_in[0];
    const float* he_ftr   = (const float*)d_in[1];
    const float* p_ftr    = (const float*)d_in[2];
    const float* q_ftr    = (const float*)d_in[3];
    const v4f*   vew1     = (const v4f*)d_in[4];
    const v4f*   vew2     = (const v4f*)d_in[5];
    // veb1/veb2 (d_in[6],d_in[7]) are derived from vew — never read.
    const float* attend_w = (const float*)d_in[8];
    const float* attend_b = (const float*)d_in[9];
    const float* align_w  = (const float*)d_in[10];
    const float* align_b  = (const float*)d_in[11];
    const float* link_w   = (const float*)d_in[12];
    const float* link_b   = (const float*)d_in[13];
    float* out = (float*)d_out;

    char* ws = (char*)d_ws;
    size_t off = 0;
    auto take = [&](size_t bytes) -> char* {
        char* p = ws + off;
        off += (bytes + 255) & ~(size_t)255;
        return p;
    };
    int*      uIdx    = (int*)take(TWO_E * 4);
    int*      vIdx    = (int*)take(TWO_E * 4);
    float*    alignB  = (float*)take(TWO_E * 4);
    float*    expAl   = (float*)take(TWO_E * 4);
    float*    nodeMax = (float*)take(N_V * 4);
    float*    nodeSum = (float*)take(N_V * 4);
    int*      nodeDeg = (int*)take(N_V * 4);
    float*    colSum  = (float*)take(MV * 4);
    _Float16* awT     = (_Float16*)take((size_t)HV * MV * 2);
    _Float16* lwT     = (_Float16*)take((size_t)KLINK * MV * 2);
    _Float16* mein    = (_Float16*)take((size_t)TWO_E * KLINK * 2);
    float*    attend  = (float*)take((size_t)TWO_E * MV * 4);
    float*    mvAcc   = (float*)take((size_t)N_V * MV * 4);

    k_init   <<<1024, 256, 0, stream>>>(mvAcc, nodeSum, nodeDeg, nodeMax, colSum);
    k_weights<<<800,  256, 0, stream>>>(attend_w, link_w, awT, lwT);
    k_extract<<<4096, 256, 0, stream>>>(vew1, vew2, uIdx, vIdx);
    k_edge   <<<TWO_E / 8, 256, 0, stream>>>(hv_ftr, he_ftr, p_ftr, q_ftr,
                                             align_w, align_b, uIdx, vIdx,
                                             mein, alignB, nodeMax, nodeDeg);
    // attend = leaky(hv_v @ attend_w + b): A = first 256 cols of mein (lda=544), K=256
    k_gemm   <<<512, 256, 0, stream>>>(mein, KLINK, awT, HV, attend_b, attend, 0);
    k_expsum <<<TWO_E / 256, 256, 0, stream>>>(alignB, vIdx, nodeMax, expAl, nodeSum);
    k_colsum <<<64, 256, 0, stream>>>(attend, colSum);
    // me_ftr = leaky(me_in @ link_w + b): K = 544 (padded), rows permuted in lwT
    k_gemm   <<<512, 256, 0, stream>>>(mein, KLINK, lwT, KLINK, link_b,
                                       out + (size_t)N_V * MV, 1);
    k_mvacc  <<<TWO_E / 8, 256, 0, stream>>>(attend, expAl, nodeSum, vIdx, mvAcc);
    k_mvfin  <<<(N_V * MV) / 256, 256, 0, stream>>>(mvAcc, nodeDeg, colSum, out);
}
